// Multi_head_attention_7627861917984
// MI455X (gfx1250) — matrix operand order repaired and checked by reading, not compiled, not run on hardware
//
#include <hip/hip_runtime.h>
#include <hip/hip_bf16.h>

typedef __attribute__((ext_vector_type(16))) _Float16 v16h;
typedef __attribute__((ext_vector_type(8)))  _Float16 v8h;
typedef __attribute__((ext_vector_type(8)))  float    v8f;
typedef __attribute__((ext_vector_type(4)))  int      v4i;

#define B_     2
#define S_     2048
#define H_     12
#define DK_    64
#define DM_    768
#define M_     (B_ * S_)     // 4096 rows for all big GEMMs

// ---------------------------------------------------------------------------
// WMMA fragment loaders (wave32, 16x16x32 f16 shapes; contiguous k order, same for A and B)
// A (16Mx32K) from row-major [*, ld]:
//   lane l: row = row0 + (l&15); kb = (l>>4)*16
//   elements 0..7  -> K = kb+e      (contiguous v8h)
//   elements 8..15 -> K = kb+8+e    (contiguous v8h)
__device__ __forceinline__ v16h load_a16(const _Float16* __restrict__ p, int ld,
                                         int row0, int k0, int lane) {
  int m  = row0 + (lane & 15);
  int kb = (lane >> 4) * 16;
  const _Float16* base = p + (size_t)m * ld + k0 + kb;
  v8h lo = *(const v8h*)(base);
  v8h hi = *(const v8h*)(base + 8);
  v16h a;
#pragma unroll
  for (int i = 0; i < 8; ++i) { a[i] = lo[i]; a[8 + i] = hi[i]; }
  return a;
}

// B (32Kx16N) supplied as Bt row-major [N][ld]  (B[k][n] == Bt[n*ld + k]):
//   lane l: col = n0 + (l&15); kb = (l>>4)*16; elements e -> K = kb+e
//   => one contiguous 32-byte v16h load per lane
__device__ __forceinline__ v16h load_bt16(const _Float16* __restrict__ bt, int ld,
                                          int n0, int k0, int lane) {
  int n  = n0 + (lane & 15);
  int kb = (lane >> 4) * 16;
  return *(const v16h*)(bt + (size_t)n * ld + k0 + kb);
}

__device__ __forceinline__ v8f wmma32(v16h a, v16h b, v8f c) {
  return __builtin_amdgcn_wmma_f32_16x16x32_f16(false, a, false, b, (short)0, c,
                                                false, false);
}

// ---------------------------------------------------------------------------
// Elementwise f32 -> f16
__global__ __launch_bounds__(256) void cvt_f32_f16(const float* __restrict__ in,
                                                   _Float16* __restrict__ out, int n) {
  int i = blockIdx.x * 256 + threadIdx.x;
  if (i < n) out[i] = (_Float16)in[i];
}

// W [K][N] f32  ->  Wt [N][K] f16
__global__ __launch_bounds__(256) void transpose_cvt(const float* __restrict__ W,
                                                     _Float16* __restrict__ Wt,
                                                     int K, int N) {
  int i = blockIdx.x * 256 + threadIdx.x;
  if (i < K * N) {
    int n = i / K, k = i - n * K;
    Wt[i] = (_Float16)W[(size_t)k * N + n];
  }
}

// ---------------------------------------------------------------------------
// WMMA GEMM, 32M x 64N tile per wave (8 accumulator chains -> WMMA pipelining):
//   C[M][N] = A[M][K](f16) * Bt[N][K](f16)^T + bias
// mode 0: fp32 row-major out
// mode 1: f16 head layout      out[((b*H + h)*S + s)*DK + d]   (Q, K)
// mode 2: f16 head-transposed  out[((b*H + h)*DK + d)*S + s]   (V)
__global__ __launch_bounds__(256)
void gemm_wmma_f16(const _Float16* __restrict__ A, const _Float16* __restrict__ Bt,
                   const float* __restrict__ bias, int M, int N, int K,
                   int mode, void* __restrict__ out) {
  int lane = threadIdx.x & 31;
  int wave = blockIdx.x * (blockDim.x >> 5) + (threadIdx.x >> 5);
  int ntiles = N >> 6;                         // 64-wide N tiles
  if (wave >= (M >> 5) * ntiles) return;
  int mTile = wave / ntiles, nTile = wave - mTile * ntiles;
  int m0 = mTile << 5, n0 = nTile << 6;

  v8f acc[2][4] = {};
  const _Float16* arow = A + (size_t)(m0 + (lane & 15)) * K;
  for (int k0 = 0; k0 < K; k0 += 32) {
    if (k0 + 64 < K) __builtin_prefetch(arow + k0 + 64, 0, 3);   // global_prefetch_b8
    v16h a0 = load_a16(A, K, m0, k0, lane);
    v16h a1 = load_a16(A, K, m0 + 16, k0, lane);
    v16h b0 = load_bt16(Bt, K, n0, k0, lane);
    v16h b1 = load_bt16(Bt, K, n0 + 16, k0, lane);
    v16h b2 = load_bt16(Bt, K, n0 + 32, k0, lane);
    v16h b3 = load_bt16(Bt, K, n0 + 48, k0, lane);
    acc[0][0] = wmma32(a0, b0, acc[0][0]);
    acc[0][1] = wmma32(a0, b1, acc[0][1]);
    acc[0][2] = wmma32(a0, b2, acc[0][2]);
    acc[0][3] = wmma32(a0, b3, acc[0][3]);
    acc[1][0] = wmma32(a1, b0, acc[1][0]);
    acc[1][1] = wmma32(a1, b1, acc[1][1]);
    acc[1][2] = wmma32(a1, b2, acc[1][2]);
    acc[1][3] = wmma32(a1, b3, acc[1][3]);
  }

  int lh = lane >> 4;
#pragma unroll
  for (int j = 0; j < 4; ++j) {
    int n = n0 + 16 * j + (lane & 15);
    float bv = bias[n];
#pragma unroll
    for (int i = 0; i < 2; ++i) {
      if (mode == 0) {
        float* o = (float*)out;
#pragma unroll
        for (int r = 0; r < 8; ++r) {
          int m = m0 + 16 * i + r + 8 * lh;
          o[(size_t)m * N + n] = acc[i][j][r] + bv;
        }
      } else if (mode == 1) {
        _Float16* o = (_Float16*)out;
        int h = n >> 6, d = n & 63;
#pragma unroll
        for (int r = 0; r < 8; ++r) {
          int m = m0 + 16 * i + r + 8 * lh;
          int bb = m >> 11, s = m & (S_ - 1);
          o[((size_t)(bb * H_ + h) * S_ + s) * DK_ + d] = (_Float16)(acc[i][j][r] + bv);
        }
      } else {
        _Float16* o = (_Float16*)out;
        int h = n >> 6, d = n & 63;
#pragma unroll
        for (int r = 0; r < 8; ++r) {
          int m = m0 + 16 * i + r + 8 * lh;
          int bb = m >> 11, s = m & (S_ - 1);
          o[((size_t)(bb * H_ + h) * DK_ + d) * S_ + s] = (_Float16)(acc[i][j][r] + bv);
        }
      }
    }
  }
}

// ---------------------------------------------------------------------------
// Flash attention, one wave per (b, h, 16-query tile), streaming 32 keys/iter.
// Scores computed transposed (ST = K * Q^T) so each lane owns one query column:
//   ST frag: lane -> q = lane&15, VGPR r -> key = chunk + r + 8*(lane>=16)
// Softmax stats are lane-local + one xor-16 shuffle. Output O^T = V^T * P.
__global__ __launch_bounds__(256)
void attn_wmma(const _Float16* __restrict__ Qh, const _Float16* __restrict__ Kh,
               const _Float16* __restrict__ Vt, const int* __restrict__ mask,
               _Float16* __restrict__ attn) {
  int lane = threadIdx.x & 31;
  int wave = blockIdx.x * 8 + (threadIdx.x >> 5);
  if (wave >= B_ * H_ * (S_ / 16)) return;
  int qTile = wave & (S_ / 16 - 1);
  int bh    = wave >> 7;               // S_/16 == 128
  int b     = bh / H_, h = bh - b * H_;
  int q0    = qTile << 4;
  int lh    = lane >> 4;

  const _Float16* Qp = Qh + (size_t)(b * H_ + h) * S_ * DK_;
  const _Float16* Kp = Kh + (size_t)(b * H_ + h) * S_ * DK_;
  const _Float16* Vp = Vt + (size_t)(b * H_ + h) * DK_ * S_;
  const int*      mp = mask + b * S_;

  // Q as B-fragments (d split 0..31 / 32..63), reused for every key chunk
  v16h bq0 = load_bt16(Qp, DK_, q0, 0, lane);
  v16h bq1 = load_bt16(Qp, DK_, q0, 32, lane);

  v8f o0 = {}, o1 = {}, o2 = {}, o3 = {};
  float m_run = -1e30f, l_run = 0.f;
  const float scale = 0.125f;  // 1/sqrt(64)

  for (int kc = 0; kc < S_; kc += 32) {
    // --- scores for two 16-key chunks (K-dim = DK = 64 -> 2 WMMAs each) ---
    v8f st0 = {}, st1 = {};
    st0 = wmma32(load_a16(Kp, DK_, kc, 0, lane), bq0, st0);
    st0 = wmma32(load_a16(Kp, DK_, kc, 32, lane), bq1, st0);
    st1 = wmma32(load_a16(Kp, DK_, kc + 16, 0, lane), bq0, st1);
    st1 = wmma32(load_a16(Kp, DK_, kc + 16, 32, lane), bq1, st1);

    // --- key mask + scale (lane rows: key = kc(+16) + 8*lh + r) ---
    v4i mm0 = *(const v4i*)(mp + kc + 8 * lh);
    v4i mm1 = *(const v4i*)(mp + kc + 8 * lh + 4);
    v4i mm2 = *(const v4i*)(mp + kc + 16 + 8 * lh);
    v4i mm3 = *(const v4i*)(mp + kc + 16 + 8 * lh + 4);
    float s0[8], s1[8];
#pragma unroll
    for (int r = 0; r < 8; ++r) {
      int k0m = (r < 4) ? mm0[r] : mm1[r - 4];
      int k1m = (r < 4) ? mm2[r] : mm3[r - 4];
      s0[r] = k0m ? st0[r] * scale : -1e9f;
      s1[r] = k1m ? st1[r] * scale : -1e9f;
    }

    // --- online softmax (per-query; lane pair combined via xor-16) ---
    float cmax = s0[0];
#pragma unroll
    for (int r = 0; r < 8; ++r) { cmax = fmaxf(cmax, s0[r]); cmax = fmaxf(cmax, s1[r]); }
    cmax = fmaxf(cmax, __shfl_xor(cmax, 16, 32));
    float m_new = fmaxf(m_run, cmax);
    float alpha = __expf(m_run - m_new);

    float p0[8], p1[8], lsum = 0.f;
#pragma unroll
    for (int r = 0; r < 8; ++r) {
      p0[r] = __expf(s0[r] - m_new);
      p1[r] = __expf(s1[r] - m_new);
      lsum += p0[r] + p1[r];
    }
    lsum += __shfl_xor(lsum, 16, 32);
    l_run = l_run * alpha + lsum;
    m_run = m_new;
#pragma unroll
    for (int r = 0; r < 8; ++r) { o0[r] *= alpha; o1[r] *= alpha; o2[r] *= alpha; o3[r] *= alpha; }

    // --- repack P from ST(C-frag) layout into B-frag layout via xor-16 ---
    // B-frag lane l needs keys (lh*16 + e) of query (l&15):
    //   lh=0: e<8 -> own p0 ; e>=8 -> partner p0
    //   lh=1: e<8 -> partner p1 ; e>=8 -> own p1
    v16h pb;
#pragma unroll
    for (int e = 0; e < 8; ++e) {
      float t0 = __shfl_xor(p0[e], 16, 32);
      float t1 = __shfl_xor(p1[e], 16, 32);
      pb[e]     = (_Float16)(lh ? t1 : p0[e]);
      pb[8 + e] = (_Float16)(lh ? p1[e] : t0);
    }

    // --- O^T += V^T(16d x 32k) * P(32k x 16q), 4 d-tiles cover DK=64 ---
    o0 = wmma32(load_a16(Vp, S_, 0,  kc, lane), pb, o0);
    o1 = wmma32(load_a16(Vp, S_, 16, kc, lane), pb, o1);
    o2 = wmma32(load_a16(Vp, S_, 32, kc, lane), pb, o2);
    o3 = wmma32(load_a16(Vp, S_, 48, kc, lane), pb, o3);
  }

  float inv = 1.0f / l_run;
  int q = lane & 15;
  _Float16* op = attn + ((size_t)(b * S_) + q0 + q) * DM_ + h * DK_;
#pragma unroll
  for (int r = 0; r < 8; ++r) {
    int d = r + 8 * lh;
    op[d]      = (_Float16)(o0[r] * inv);
    op[16 + d] = (_Float16)(o1[r] * inv);
    op[32 + d] = (_Float16)(o2[r] * inv);
    op[48 + d] = (_Float16)(o3[r] * inv);
  }
}

// ---------------------------------------------------------------------------
extern "C" void kernel_launch(void* const* d_in, const int* in_sizes, int n_in,
                              void* d_out, int out_size, void* d_ws, size_t ws_size,
                              hipStream_t stream) {
  const float* q  = (const float*)d_in[0];
  const float* k  = (const float*)d_in[1];
  const float* v  = (const float*)d_in[2];
  const float* Wq = (const float*)d_in[3];
  const float* bq = (const float*)d_in[4];
  const float* Wk = (const float*)d_in[5];
  const float* bk = (const float*)d_in[6];
  const float* Wv = (const float*)d_in[7];
  const float* bv = (const float*)d_in[8];
  const float* Wo = (const float*)d_in[9];
  const float* bo = (const float*)d_in[10];
  const int*   mask = (const int*)d_in[11];

  // workspace carving (all offsets 256B-aligned)
  const size_t X16  = (size_t)M_ * DM_ * sizeof(_Float16);  // 6,291,456 B
  const size_t W16  = (size_t)DM_ * DM_ * sizeof(_Float16); // 1,179,648 B
  char* ws = (char*)d_ws;
  _Float16* q16    = (_Float16*)(ws);
  _Float16* k16    = (_Float16*)(ws + X16);
  _Float16* v16    = (_Float16*)(ws + 2 * X16);
  _Float16* attn16 = (_Float16*)(ws + 3 * X16);
  _Float16* Wqt    = (_Float16*)(ws + 4 * X16);
  _Float16* Wkt    = (_Float16*)(ws + 4 * X16 + W16);
  _Float16* Wvt    = (_Float16*)(ws + 4 * X16 + 2 * W16);
  _Float16* Wot    = (_Float16*)(ws + 4 * X16 + 3 * W16);
  _Float16* Qh     = (_Float16*)(ws + 4 * X16 + 4 * W16);
  _Float16* Kh     = (_Float16*)(ws + 5 * X16 + 4 * W16);
  _Float16* Vt     = (_Float16*)(ws + 6 * X16 + 4 * W16);

  const int nX = M_ * DM_;
  cvt_f32_f16<<<nX / 256, 256, 0, stream>>>(q, q16, nX);
  cvt_f32_f16<<<nX / 256, 256, 0, stream>>>(k, k16, nX);
  cvt_f32_f16<<<nX / 256, 256, 0, stream>>>(v, v16, nX);

  const int nW = DM_ * DM_;
  transpose_cvt<<<nW / 256, 256, 0, stream>>>(Wq, Wqt, DM_, DM_);
  transpose_cvt<<<nW / 256, 256, 0, stream>>>(Wk, Wkt, DM_, DM_);
  transpose_cvt<<<nW / 256, 256, 0, stream>>>(Wv, Wvt, DM_, DM_);
  transpose_cvt<<<nW / 256, 256, 0, stream>>>(Wo, Wot, DM_, DM_);

  const int gemmBlocks = (M_ / 32) * (DM_ / 64) / 8;  // 192 blocks of 8 waves
  gemm_wmma_f16<<<gemmBlocks, 256, 0, stream>>>(q16, Wqt, bq, M_, DM_, DM_, 1, Qh);
  gemm_wmma_f16<<<gemmBlocks, 256, 0, stream>>>(k16, Wkt, bk, M_, DM_, DM_, 1, Kh);
  gemm_wmma_f16<<<gemmBlocks, 256, 0, stream>>>(v16, Wvt, bv, M_, DM_, DM_, 2, Vt);

  const int attnBlocks = B_ * H_ * (S_ / 16) / 8;     // 384
  attn_wmma<<<attnBlocks, 256, 0, stream>>>(Qh, Kh, Vt, mask, attn16);

  gemm_wmma_f16<<<gemmBlocks, 256, 0, stream>>>(attn16, Wot, bo, M_, DM_, DM_, 0, d_out);
}
